// ProjectionMatcher_65317862638090
// MI455X (gfx1250) — matrix operand order repaired and checked by reading, not compiled, not run on hardware
//
#include <hip/hip_runtime.h>
#include <hip/hip_bf16.h>
#include <math.h>

#define D 128
#define DH 65          // D/2+1
#define NB 8
#define NOPT 125
#define TWO_PI 6.28318530717958647692f

typedef __attribute__((ext_vector_type(16))) _Float16 v16h;
typedef __attribute__((ext_vector_type(8)))  _Float16 v8h;
typedef __attribute__((ext_vector_type(8)))  float    v8f;

// ---------------------------------------------------------------------------
// Euler ZYZ -> rotation matrix (matches reference formulas)
// ---------------------------------------------------------------------------
__device__ __forceinline__ void euler_zyz_to_R(float a, float b, float c, float* R) {
  float sa, ca, sb, cb, sc, cc;
  __sincosf(a, &sa, &ca);
  __sincosf(b, &sb, &cb);
  __sincosf(c, &sc, &cc);
  R[0] = ca * cb * cc - sa * sc; R[1] = -ca * cb * sc - sa * cc; R[2] = ca * sb;
  R[3] = sa * cb * cc + ca * sc; R[4] = -sa * cb * sc + ca * cc; R[5] = sa * sb;
  R[6] = -sb * cc;               R[7] = sb * sc;                 R[8] = cb;
}

// ---------------------------------------------------------------------------
// WMMA fragment loaders (16x16x32 f16, wave32; contiguous k order, same in A and B)
//
// A (16x32): lane m = lane&15; elem j -> K = k0 + 16*hlf + j
//   Row-major source => one contiguous 32B run => 2x b128 loads.
//
// B (32x16): lane n = lane&15; elem j -> K = k0 + 16*hlf + j
//   Source is PRE-SWIZZLED: per (K-chunk, N-tile) block, each column's 32
//   K-values contiguous => one aligned 32B vector load (2x b128).
//   swizzled index = ((blk*16 + col)*32 + koff),  blk = kc*NTILES + nt
// ---------------------------------------------------------------------------
__device__ __forceinline__ v16h fragA16(const _Float16* base, int row, int stride,
                                        int k0, int hlf) {
  const _Float16* p = base + row * stride + k0 + 16 * hlf;
  v8h lo = *(const v8h*)p;
  v8h hi = *(const v8h*)(p + 8);
  return __builtin_shufflevector(lo, hi, 0, 1, 2, 3, 4, 5, 6, 7,
                                 8, 9, 10, 11, 12, 13, 14, 15);
}

__device__ __forceinline__ v16h fragBsw(const _Float16* base, int blk, int col,
                                        int hlf) {
  return *(const v16h*)(base + (blk * 16 + col) * 32 + 16 * hlf);
}

// ---------------------------------------------------------------------------
// Kernel 0: build f16 DFT basis tables.
//   Wy[y][r]    = exp(+2*pi*i*(r+64)*y/128)  row-major (A operand)
//   WyIneg      = -Im(Wy)  (lets the complex GEMM run with zero runtime
//                 sign flips: Re += WyR*PR + WyIneg*PI)
//   Cx/Sx: Hermitian-folded x-basis, stored B-swizzled with NTILES=8,
//   K padded 65->96 with zeros.
// ---------------------------------------------------------------------------
__global__ void build_tables_k(_Float16* WyR, _Float16* WyI, _Float16* WyIneg,
                               _Float16* Cx, _Float16* Sx) {
  int t = blockIdx.x * blockDim.x + threadIdx.x;
  if (t < D * D) {
    int y = t / D, r = t % D;
    int p = ((r + 64) * y) & 127;
    float s, c;
    __sincosf(TWO_PI * (float)p / 128.0f, &s, &c);
    WyR[t] = (_Float16)c;
    WyI[t] = (_Float16)s;
    WyIneg[t] = (_Float16)(-s);
  }
  if (t < 96 * D) {
    int k = t / D, x = t % D;
    float cr = 0.0f, si = 0.0f;
    if (k < DH) {
      float ck = (k == 0 || k == 64) ? 1.0f : 2.0f;
      float s, c;
      __sincosf(TWO_PI * (float)((k * x) & 127) / 128.0f, &s, &c);
      cr = ck * c;
      si = -ck * s;
    }
    int kc = k >> 5, koff = k & 31, nt = x >> 4, col = x & 15;
    int sw = ((kc * 8 + nt) * 16 + col) * 32 + koff;
    Cx[sw] = (_Float16)cr;
    Sx[sw] = (_Float16)si;
  }
}

// ---------------------------------------------------------------------------
// Volume rFFT pass 1 (x axis, real->complex). Spatial fftshift folded into
// read indices on all three axes. out[z][y][kx], kx in [0,65).
// ---------------------------------------------------------------------------
__global__ void vol_fft_x_k(const float* __restrict__ vol, float* __restrict__ out) {
  __shared__ float row[D];
  __shared__ float twc[D], tws[D];
  int zy = blockIdx.x;          // z*128 + y
  int tid = threadIdx.x;        // 128 threads
  int z = zy >> 7, y = zy & 127;
  int zz = (z + 64) & 127, yy = (y + 64) & 127;
  row[tid] = vol[(zz * D + yy) * D + ((tid + 64) & 127)];
  float s, c;
  __sincosf(TWO_PI * (float)tid / 128.0f, &s, &c);
  twc[tid] = c; tws[tid] = -s;  // exp(-i*2*pi*tid/128)
  __syncthreads();
  if (tid < DH) {
    float re = 0.0f, im = 0.0f;
    for (int x = 0; x < D; ++x) {
      int p = (tid * x) & 127;
      re += row[x] * twc[p];
      im += row[x] * tws[p];
    }
    int o = (zy * DH + tid) * 2;
    out[o] = re; out[o + 1] = im;
  }
}

// ---------------------------------------------------------------------------
// Complex DFT along the "y" (middle) axis, fftshift of output rows.
// in/out layout [z][y][kx] (z = batch: 128 for volume, 8 for images).
// ---------------------------------------------------------------------------
__global__ void fft_cols_y_k(const float* __restrict__ in, float* __restrict__ out) {
  __shared__ float cre[D], cim[D], twc[D], tws[D];
  int blk = blockIdx.x;          // z*DH + kx
  int tid = threadIdx.x;         // 128 threads: ky
  int z = blk / DH, kx = blk % DH;
  int idx = ((z * D + tid) * DH + kx) * 2;
  cre[tid] = in[idx]; cim[tid] = in[idx + 1];
  float s, c;
  __sincosf(TWO_PI * (float)tid / 128.0f, &s, &c);
  twc[tid] = c; tws[tid] = -s;
  __syncthreads();
  float re = 0.0f, im = 0.0f;
  for (int y = 0; y < D; ++y) {
    int p = (tid * y) & 127;
    float wr = twc[p], wi = tws[p];
    re += cre[y] * wr - cim[y] * wi;
    im += cre[y] * wi + cim[y] * wr;
  }
  int rs = (tid + 64) & 127;     // fftshift ky
  int o = ((z * D + rs) * DH + kx) * 2;
  out[o] = re; out[o + 1] = im;
}

// ---------------------------------------------------------------------------
// Volume DFT along z axis, fftshift of output planes. in [z][ky][kx].
// ---------------------------------------------------------------------------
__global__ void vol_fft_z_k(const float* __restrict__ in, float* __restrict__ out) {
  __shared__ float cre[D], cim[D], twc[D], tws[D];
  int blk = blockIdx.x;          // ky*DH + kx
  int tid = threadIdx.x;         // kz
  int ky = blk / DH, kx = blk % DH;
  int idx = ((tid * D + ky) * DH + kx) * 2;
  cre[tid] = in[idx]; cim[tid] = in[idx + 1];
  float s, c;
  __sincosf(TWO_PI * (float)tid / 128.0f, &s, &c);
  twc[tid] = c; tws[tid] = -s;
  __syncthreads();
  float re = 0.0f, im = 0.0f;
  for (int zz = 0; zz < D; ++zz) {
    int p = (tid * zz) & 127;
    float wr = twc[p], wi = tws[p];
    re += cre[zz] * wr - cim[zz] * wi;
    im += cre[zz] * wi + cim[zz] * wr;
  }
  int rs = (tid + 64) & 127;     // fftshift kz
  int o = ((rs * D + ky) * DH + kx) * 2;
  out[o] = re; out[o + 1] = im;
}

// ---------------------------------------------------------------------------
// Image rFFT pass 1 (x axis) with circular mask; spatial fftshift folded in.
// out [b][y][kx].
// ---------------------------------------------------------------------------
__global__ void img_fft_x_k(const float* __restrict__ imgs, float* __restrict__ out) {
  __shared__ float row[D], twc[D], tws[D];
  int by = blockIdx.x;          // b*128 + y
  int tid = threadIdx.x;
  int b = by >> 7, y = by & 127;
  int yo = (y + 64) & 127;
  int xo = (tid + 64) & 127;
  float dy = (float)(yo - 64), dx = (float)(xo - 64);
  float m = (dy * dy + dx * dx <= 4096.0f) ? 1.0f : 0.0f;
  row[tid] = imgs[(b * D + yo) * D + xo] * m;
  float s, c;
  __sincosf(TWO_PI * (float)tid / 128.0f, &s, &c);
  twc[tid] = c; tws[tid] = -s;
  __syncthreads();
  if (tid < DH) {
    float re = 0.0f, im = 0.0f;
    for (int x = 0; x < D; ++x) {
      int p = (tid * x) & 127;
      re += row[x] * twc[p];
      im += row[x] * tws[p];
    }
    int o = (by * DH + tid) * 2;
    out[o] = re; out[o + 1] = im;
  }
}

__device__ __forceinline__ int selmap(int i) { return i + ((i & 2) ? 4 : 0); } // {0,1,6,7}

// ---------------------------------------------------------------------------
// Main kernel: one workgroup per (b, opt). 256 threads = 8 waves.
// Phase 1: slice gather + CTF + cross-spectrum -> prod (f16 LDS, B-swizzled,
//          pre-scaled by 1/D^2)
// Stage 1: tmp = Wy * prod (complex GEMM, WMMA f16->f32); wave w owns M-row w,
//          A fragments (WyR/WyI/WyIneg) hoisted across the 5 N tiles; no
//          runtime sign flips.
// Stage 2: corr window tiles = Re(tmp)*Cx + Im(tmp)*Sx (WMMA); wave w owns
//          M-row selmap(w>>1) and N pair {0,1} or {6,7}; argmax reduction.
// ---------------------------------------------------------------------------
__global__ void __launch_bounds__(256)
match_k(const float* __restrict__ volF, const float* __restrict__ fparts,
        const float* __restrict__ ctfs, const float* __restrict__ rotmats,
        const _Float16* __restrict__ WyR, const _Float16* __restrict__ WyI,
        const _Float16* __restrict__ WyIneg,
        const _Float16* __restrict__ Cx, const _Float16* __restrict__ Sx,
        float* __restrict__ maxc, int* __restrict__ srow, int* __restrict__ scol) {
  extern __shared__ char smem[];
  _Float16* prodR = (_Float16*)smem;          // B-swizzled [4 kc][5 nt][16][32]
  _Float16* prodI = prodR + D * 80;
  _Float16* tmpR  = prodI + D * 80;           // row-major [128][96]
  _Float16* tmpI  = tmpR + D * 96;
  float* coef = (float*)(tmpI + D * 96);      // 8 floats
  float* redV = coef + 8;                     // 256
  int*   redI = (int*)(redV + 256);           // 256

  int bo = blockIdx.x;
  int b = bo / NOPT, o = bo % NOPT;
  int tid = threadIdx.x;

  // prefetch stage-2 basis tables (global_prefetch_b8)
  if (tid < 96) {
    __builtin_prefetch(Cx + tid * D, 0, 3);
    __builtin_prefetch(Sx + tid * D, 0, 3);
  }

  if (tid == 0) {
    const float* Rm = rotmats + b * 9;
    float r22 = fminf(1.0f, fmaxf(-1.0f, Rm[8]));
    float eb = acosf(r22);
    float ea = atan2f(Rm[5], Rm[2]);
    float ec = atan2f(Rm[7], -Rm[6]);
    const float R2DG = 57.29577951308232f, DG2R = 0.017453292519943295f;
    int i0 = o / 25, i1 = (o / 5) % 5, i2 = o % 5;
    float e0 = ea * R2DG + (-6.0f + 3.0f * (float)i0);
    float e1 = eb * R2DG + (-6.0f + 3.0f * (float)i1);
    float e2 = ec * R2DG + (-6.0f + 3.0f * (float)i2);
    float R[9];
    euler_zyz_to_R(e0 * DG2R, e1 * DG2R, e2 * DG2R, R);
    // coords_i = Rz[i][1]*gy + Rz[i][2]*gx, Rz[i][j] = R[2-i][2-j]
    coef[0] = R[(2 - 0) * 3 + 1]; coef[1] = R[(2 - 1) * 3 + 1]; coef[2] = R[(2 - 2) * 3 + 1];
    coef[3] = R[(2 - 0) * 3 + 0]; coef[4] = R[(2 - 1) * 3 + 0]; coef[5] = R[(2 - 2) * 3 + 0];
  }
  __syncthreads();
  float Ay0 = coef[0], Ay1 = coef[1], Ay2 = coef[2];
  float Ax0 = coef[3], Ax1 = coef[4], Ax2 = coef[5];

  const float invDD = 1.0f / 16384.0f;   // irfft2 normalization, applied early

  // ---- phase 1: central slice gather + CTF + cross spectrum -> LDS f16 ----
  for (int e = tid; e < D * 80; e += 256) {
    int r = e / 80, kx = e % 80;          // r = K index (ky), kx = N index
    float pre = 0.0f, pim = 0.0f;
    if (kx < DH) {
      float gy = (float)(r - 64) / 128.0f;
      float gx = (float)kx / 128.0f;
      float c0 = Ay0 * gy + Ax0 * gx;
      float c1 = Ay1 * gy + Ax1 * gx;
      float c2 = Ay2 * gy + Ax2 * gx;
      bool cj = (c2 < 0.0f);
      if (cj) { c0 = -c0; c1 = -c1; c2 = -c2; }
      float fz = c0 * 128.0f + 64.0f;
      float fy = c1 * 128.0f + 64.0f;
      float fx = c2 * 128.0f;
      float flz = floorf(fz), fly = floorf(fy), flx = floorf(fx);
      int iz = (int)flz, iy = (int)fly, ix = (int)flx;
      float tz = fz - flz, ty = fy - fly, tx = fx - flx;
      float sre = 0.0f, sim = 0.0f;
#pragma unroll
      for (int dz = 0; dz < 2; ++dz)
#pragma unroll
        for (int dy2 = 0; dy2 < 2; ++dy2)
#pragma unroll
          for (int dx2 = 0; dx2 < 2; ++dx2) {
            int z = iz + dz, y = iy + dy2, x = ix + dx2;
            if (z >= 0 && z < D && y >= 0 && y < D && x >= 0 && x < DH) {
              float w = (dz ? tz : 1.0f - tz) * (dy2 ? ty : 1.0f - ty) *
                        (dx2 ? tx : 1.0f - tx);
              int id = ((z * D + y) * DH + x) * 2;
              sre += w * volF[id];
              sim += w * volF[id + 1];
            }
          }
      if (cj) sim = -sim;
      float ctf = ctfs[(b * D + r) * DH + kx];
      sre *= ctf; sim *= ctf;
      int fi = ((b * D + r) * DH + kx) * 2;
      float fr = fparts[fi], fim = fparts[fi + 1];
      // prod = fpart * conj(proj), scaled by 1/D^2 to keep f16 in range
      pre = (fr * sre + fim * sim) * invDD;
      pim = (fim * sre - fr * sim) * invDD;
    }
    // B-swizzled store: blk = (r/32)*5 + kx/16
    int sw = (((r >> 5) * 5 + (kx >> 4)) * 16 + (kx & 15)) * 32 + (r & 31);
    prodR[sw] = (_Float16)pre;
    prodI[sw] = (_Float16)pim;
  }
  // zero K-padding columns 80..95 of tmp (stage-2 reads K up to 96)
  for (int e = tid; e < D * 16; e += 256) {
    int r = e / 16, cp = 80 + (e % 16);
    tmpR[r * 96 + cp] = (_Float16)0.0f;
    tmpI[r * 96 + cp] = (_Float16)0.0f;
  }
  __syncthreads();

  int wid = tid >> 5, lane = tid & 31, hlf = lane >> 4, lcol = lane & 15;

  // ---- stage 1: tmp[y][kx] = sum_r Wy[y][r] * prod[r][kx] (complex GEMM) ----
  // Re(tmp) = sum WyR*PR + WyIneg*PI ; Im(tmp) = sum WyR*PI + WyI*PR
  {
    int mrow = wid * 16 + lcol;
    v16h aR[4], aI[4], aN[4];
#pragma unroll
    for (int kc = 0; kc < 4; ++kc) {
      aR[kc] = fragA16(WyR, mrow, D, kc * 32, hlf);
      aI[kc] = fragA16(WyI, mrow, D, kc * 32, hlf);
      aN[kc] = fragA16(WyIneg, mrow, D, kc * 32, hlf);
    }
    for (int nt = 0; nt < 5; ++nt) {
      v8f accR = {};
      v8f accI = {};
#pragma unroll
      for (int kc = 0; kc < 4; ++kc) {
        v16h bR = fragBsw(prodR, kc * 5 + nt, lcol, hlf);
        v16h bI = fragBsw(prodI, kc * 5 + nt, lcol, hlf);
        accR = __builtin_amdgcn_wmma_f32_16x16x32_f16(false, aR[kc], false, bR, (short)0, accR, false, false);
        accR = __builtin_amdgcn_wmma_f32_16x16x32_f16(false, aN[kc], false, bI, (short)0, accR, false, false);
        accI = __builtin_amdgcn_wmma_f32_16x16x32_f16(false, aR[kc], false, bI, (short)0, accI, false, false);
        accI = __builtin_amdgcn_wmma_f32_16x16x32_f16(false, aI[kc], false, bR, (short)0, accI, false, false);
      }
#pragma unroll
      for (int rr = 0; rr < 8; ++rr) {
        int m = wid * 16 + rr + 8 * hlf;
        int n = nt * 16 + lcol;
        tmpR[m * 96 + n] = (_Float16)accR[rr];
        tmpI[m * 96 + n] = (_Float16)accI[rr];
      }
    }
  }
  __syncthreads();

  // ---- stage 2: corr = Re(tmp)*Cx + Im(tmp)*Sx, only fftshift-window tiles --
  // Needed rows/cols (unshifted): 0..31, 96..127 -> tiles {0,1,6,7}x{0,1,6,7}.
  // Wave w: M tile selmap(w>>1), N tiles {0,1} (w even) or {6,7} (w odd).
  float bestV = -3.4e38f;
  int bestI = 1 << 30;
  {
    int mt = selmap(wid >> 1);
    int nbase = (wid & 1) ? 6 : 0;
    int mrow = mt * 16 + lcol;
    v16h aR[3], aI[3];
#pragma unroll
    for (int kc = 0; kc < 3; ++kc) {
      aR[kc] = fragA16(tmpR, mrow, 96, kc * 32, hlf);
      aI[kc] = fragA16(tmpI, mrow, 96, kc * 32, hlf);
    }
#pragma unroll
    for (int tn = 0; tn < 2; ++tn) {
      int nt = nbase + tn;
      v8f acc = {};
#pragma unroll
      for (int kc = 0; kc < 3; ++kc) {
        v16h bC = fragBsw(Cx, kc * 8 + nt, lcol, hlf);
        v16h bS = fragBsw(Sx, kc * 8 + nt, lcol, hlf);
        acc = __builtin_amdgcn_wmma_f32_16x16x32_f16(false, aR[kc], false, bC, (short)0, acc, false, false);
        acc = __builtin_amdgcn_wmma_f32_16x16x32_f16(false, aI[kc], false, bS, (short)0, acc, false, false);
      }
#pragma unroll
      for (int rr = 0; rr < 8; ++rr) {
        int m = mt * 16 + rr + 8 * hlf;
        int n = nt * 16 + lcol;
        float v = acc[rr];
        int ys = (m + 64) & 127, xs = (n + 64) & 127;   // fftshift
        int widx = (ys - 32) * 64 + (xs - 32);          // flat window index
        if (v > bestV || (v == bestV && widx < bestI)) { bestV = v; bestI = widx; }
      }
    }
  }
  redV[tid] = bestV;
  redI[tid] = bestI;
  __syncthreads();
  for (int s = 128; s > 0; s >>= 1) {
    if (tid < s) {
      float v2 = redV[tid + s];
      int i2 = redI[tid + s];
      if (v2 > redV[tid] || (v2 == redV[tid] && i2 < redI[tid])) {
        redV[tid] = v2; redI[tid] = i2;
      }
    }
    __syncthreads();
  }
  if (tid == 0) {
    maxc[bo] = redV[0];
    srow[bo] = redI[0] / 64 + 32;
    scol[bo] = redI[0] % 64 + 32;
  }
}

// ---------------------------------------------------------------------------
// Final reduction: per image top-1 over 125 options + stats + outputs.
// d_out: top_v[8] | rotmats[8*9] | shifts[8*2] | weight[8]  (104 floats)
// ---------------------------------------------------------------------------
__global__ void finalize_k(const float* __restrict__ maxc, const int* __restrict__ srow,
                           const int* __restrict__ scol, const float* __restrict__ rotmats,
                           float* __restrict__ out) {
  int b = blockIdx.x;
  if (threadIdx.x != 0) return;
  float sum = 0.0f, sumsq = 0.0f, best = -3.4e38f;
  int bi = 0;
  for (int o = 0; o < NOPT; ++o) {
    float v = maxc[b * NOPT + o];
    sum += v; sumsq += v * v;
    if (v > best) { best = v; bi = o; }
  }
  float mean = sum / 125.0f;
  float var = fmaxf((sumsq - sum * sum / 125.0f) / 124.0f, 0.0f);
  float sd = sqrtf(var);
  float w = 0.5f * (1.0f + erff((best - mean) / ((sd + 1e-6f) * 1.4142135623730951f)));

  const float* Rm = rotmats + b * 9;
  float r22 = fminf(1.0f, fmaxf(-1.0f, Rm[8]));
  float eb = acosf(r22);
  float ea = atan2f(Rm[5], Rm[2]);
  float ec = atan2f(Rm[7], -Rm[6]);
  const float R2DG = 57.29577951308232f, DG2R = 0.017453292519943295f;
  int i0 = bi / 25, i1 = (bi / 5) % 5, i2 = bi % 5;
  float e0 = ea * R2DG + (-6.0f + 3.0f * (float)i0);
  float e1 = eb * R2DG + (-6.0f + 3.0f * (float)i1);
  float e2 = ec * R2DG + (-6.0f + 3.0f * (float)i2);
  float R[9];
  euler_zyz_to_R(e0 * DG2R, e1 * DG2R, e2 * DG2R, R);

  out[b] = best;
#pragma unroll
  for (int j = 0; j < 9; ++j) out[8 + b * 9 + j] = R[j];
  float col = (float)scol[b * NOPT + bi];
  float row = (float)srow[b * NOPT + bi];
  out[80 + b * 2 + 0] = -(col - 64.0f) * 1.5f;   // PIXEL_SIZE = 1.5
  out[80 + b * 2 + 1] = -(row - 64.0f) * 1.5f;
  out[96 + b] = w;
}

// ---------------------------------------------------------------------------
extern "C" void kernel_launch(void* const* d_in, const int* in_sizes, int n_in,
                              void* d_out, int out_size, void* d_ws, size_t ws_size,
                              hipStream_t stream) {
  (void)in_sizes; (void)n_in; (void)out_size; (void)ws_size;
  const float* vol = (const float*)d_in[0];
  const float* imgs = (const float*)d_in[1];
  const float* ctfs = (const float*)d_in[2];
  const float* rotmats = (const float*)d_in[3];
  float* out = (float*)d_out;

  char* ws = (char*)d_ws;
  size_t off = 0;
  auto wsalloc = [&](size_t bytes) -> char* {
    char* p = ws + off;
    off += (bytes + 255) & ~(size_t)255;
    return p;
  };
  float* bufA = (float*)wsalloc((size_t)D * D * DH * 2 * sizeof(float)); // 8.5 MB
  float* bufB = (float*)wsalloc((size_t)D * D * DH * 2 * sizeof(float)); // 8.5 MB
  float* fparts = (float*)wsalloc((size_t)NB * D * DH * 2 * sizeof(float));
  _Float16* WyR = (_Float16*)wsalloc((size_t)D * D * sizeof(_Float16));
  _Float16* WyI = (_Float16*)wsalloc((size_t)D * D * sizeof(_Float16));
  _Float16* WyIneg = (_Float16*)wsalloc((size_t)D * D * sizeof(_Float16));
  _Float16* Cx  = (_Float16*)wsalloc((size_t)96 * D * sizeof(_Float16));
  _Float16* Sx  = (_Float16*)wsalloc((size_t)96 * D * sizeof(_Float16));
  float* maxc = (float*)wsalloc((size_t)NB * NOPT * sizeof(float));
  int* srow = (int*)wsalloc((size_t)NB * NOPT * sizeof(int));
  int* scol = (int*)wsalloc((size_t)NB * NOPT * sizeof(int));

  // basis tables
  build_tables_k<<<(D * D + 255) / 256, 256, 0, stream>>>(WyR, WyI, WyIneg, Cx, Sx);
  // volume 3D rFFT (ping-pong: bufA -> bufB -> bufA); volF = bufA
  vol_fft_x_k<<<D * D, D, 0, stream>>>(vol, bufA);
  fft_cols_y_k<<<D * DH, D, 0, stream>>>(bufA, bufB);
  vol_fft_z_k<<<D * DH, D, 0, stream>>>(bufB, bufA);
  const float* volF = bufA;
  // image 2D rFFTs (bufB reused as pass-1 scratch)
  img_fft_x_k<<<NB * D, D, 0, stream>>>(imgs, bufB);
  fft_cols_y_k<<<NB * DH, D, 0, stream>>>(bufB, fparts);
  // main matcher: 1000 workgroups, 8 waves each, ~92 KB LDS
  size_t ldsMain = (size_t)(2 * (D * 80) + 2 * (D * 96)) * sizeof(_Float16) +
                   (8 + 256) * sizeof(float) + 256 * sizeof(int);
  match_k<<<NB * NOPT, 256, ldsMain, stream>>>(volF, fparts, ctfs, rotmats,
                                               WyR, WyI, WyIneg, Cx, Sx,
                                               maxc, srow, scol);
  finalize_k<<<NB, 32, 0, stream>>>(maxc, srow, scol, rotmats, out);
}
